// LSTMv2_3229815407363
// MI455X (gfx1250) — matrix operand order repaired and checked by reading, not compiled, not run on hardware
//
#include <hip/hip_runtime.h>

#define T_SEQ 2048
#define HID   1024
#define NLAY  4
#define NWG   32
#define TPB   256
#define KH    512   // K split in half per thread in the scan

typedef unsigned short u16;
typedef unsigned int   u32;

typedef __attribute__((ext_vector_type(16))) __bf16 v16bf;
typedef __attribute__((ext_vector_type(8)))  float  v8f;

__device__ __forceinline__ float bflo(u32 u) { return __uint_as_float(u << 16); }
__device__ __forceinline__ float bfhi(u32 u) { return __uint_as_float(u & 0xffff0000u); }

__device__ __forceinline__ u16 f2bf(float f) {
  u32 u = __float_as_uint(f);
  u32 r = u + 0x7fffu + ((u >> 16) & 1u);   // round-to-nearest-even
  return (u16)(r >> 16);
}

// ---------------- per-call init: zero sync + state ----------------
__global__ void k_init(u32* syncp, float* hstate, float* sstate) {
  int i = blockIdx.x * blockDim.x + threadIdx.x;
  if (i < 64) syncp[i] = 0u;
  if (i < NLAY * HID) { hstate[i] = 0.f; sstate[i] = 0.f; }
}

// ------- fp32 -> bf16 conversion with [layer][gate] interleave -------
__global__ void k_cvt(const float* __restrict__ src, u16* __restrict__ dst,
                      int gate, int nmat, int matsize) {
  long long i = (long long)blockIdx.x * blockDim.x + threadIdx.x;
  long long total = (long long)nmat * matsize;
  if (i >= total) return;
  int l = (int)(i / matsize);
  long long j = i - (long long)l * matsize;
  dst[((long long)(l * 4 + gate)) * matsize + j] = f2bf(src[i]);
}

// ---------------- Phase 1: Z0[g][t][r] = U4[g] @ x^T via WMMA bf16 ----------------
// A = U4 (M = 4*1024 rows, K = 1024, row-major bf16)
// B = xbf (columns t are rows of x, K-contiguous)
__global__ void __launch_bounds__(TPB, 1)
k_z0gemm(const u16* __restrict__ A, const u16* __restrict__ Bx, float* __restrict__ Z0) {
  const int lane = threadIdx.x & 31;
  const int wave = threadIdx.x >> 5;
  const int tile = blockIdx.x * 8 + wave;    // 256 Mtiles x 128 Ntiles = 32768 tiles
  const int mtile = tile & 255;
  const int ntile = tile >> 8;
  const int row0 = mtile << 4;
  const int t0   = ntile << 4;
  const int l15  = lane & 15;
  const bool hi  = lane >= 16;

  // A layout (16-bit 16x32): lane<16 holds K {k0..k0+7, k0+16..k0+23}; lane>=16 offset +8
  const u16* ap = A  + (size_t)(row0 + l15) * HID + (hi ? 8 : 0);
  // B layout (32x16): lane holds column t0+l15, K {k0..k0+7, k0+16..k0+23} (lanes>=16 offset +8)
  const u16* bp = Bx + (size_t)(t0  + l15) * HID + (hi ? 8 : 0);

  v8f c = {};
  #pragma unroll 4
  for (int k0 = 0; k0 < HID; k0 += 32) {
    union { uint4 q[2]; v16bf v; } a, b;
    a.q[0] = *(const uint4*)(ap + k0);        // K base..base+7
    a.q[1] = *(const uint4*)(ap + k0 + 16);   // K base+16..base+23
    b.q[0] = *(const uint4*)(bp + k0);        // K base..base+7
    b.q[1] = *(const uint4*)(bp + k0 + 16);   // K base+16..base+23
    c = __builtin_amdgcn_wmma_f32_16x16x32_bf16(false, a.v, false, b.v,
                                                (short)0, c, false, false);
  }

  // D layout: lane column N = l15, rows M = i + (hi?8:0) -> 8 consecutive output rows
  const int tcol = t0 + l15;
  const int m0   = row0 + (hi ? 8 : 0);
  const int g    = m0 >> 10;
  const int r0   = m0 & 1023;
  float* out = Z0 + ((size_t)g * T_SEQ + tcol) * HID + r0;
  union { v8f v; float4 f[2]; } cu; cu.v = c;
  *(float4*)(out)     = cu.f[0];
  *(float4*)(out + 4) = cu.f[1];
}

// ---------------- Phase 2: sequential scan (persistent kernel) ----------------
__device__ __forceinline__ float dot512(const u16* __restrict__ w,
                                        const float* __restrict__ hv) {
  float acc = 0.f;
  #pragma unroll 8
  for (int k = 0; k < KH; k += 8) {
    uint4 wq = *(const uint4*)(w + k);          // 8 bf16 weights
    float4 h0 = *(const float4*)(hv + k);       // LDS b128 loads
    float4 h1 = *(const float4*)(hv + k + 4);
    acc += bflo(wq.x) * h0.x + bfhi(wq.x) * h0.y
         + bflo(wq.y) * h0.z + bfhi(wq.y) * h0.w
         + bflo(wq.z) * h1.x + bfhi(wq.z) * h1.y
         + bflo(wq.w) * h1.z + bfhi(wq.w) * h1.w;
  }
  return acc;
}

__global__ void __launch_bounds__(TPB, 1)
k_scan(const u16* __restrict__ W4, const u16* __restrict__ UH4,
       const float* __restrict__ Z0,
       const float* __restrict__ Bf, const float* __restrict__ Bg,
       const float* __restrict__ Bq, const float* __restrict__ Bc,
       float* __restrict__ hstate, float* __restrict__ sstate,
       u32* syncp)
{
  __shared__ __align__(16) float h_rec[HID];   // h[l] from t-1
  __shared__ __align__(16) float h_low[HID];   // h[l-1] from t
  __shared__ float partial[TPB];
  __shared__ float pre[32][4];

  const int tid   = threadIdx.x;
  const int wg    = blockIdx.x;
  const int khalf = tid & 1;
  const int gate  = (tid >> 1) & 3;
  const int rlo   = tid >> 3;                  // 0..31
  const int row   = wg * 32 + rlo;             // 0..1023
  const int kb    = khalf * KH;

  u32* cnt = syncp;
  u32* gen = syncp + 16;

  const float* bias = (gate == 0) ? Bf : (gate == 1) ? Bg : (gate == 2) ? Bq : Bc;

  for (int t = 0; t < T_SEQ; ++t) {
    for (int l = 0; l < NLAY; ++l) {
      for (int i = tid; i < HID; i += TPB) {
        h_rec[i] = hstate[l * HID + i];
        if (l > 0) h_low[i] = hstate[(l - 1) * HID + i];
      }
      __syncthreads();

      const u16* wrow = W4 + ((size_t)((l * 4 + gate) * HID + row)) * HID + kb;
      float acc = dot512(wrow, h_rec + kb);
      if (l > 0) {
        const u16* urow = UH4 + ((size_t)(((l - 1) * 4 + gate) * HID + row)) * HID + kb;
        acc += dot512(urow, h_low + kb);
      }
      partial[tid] = acc;
      __syncthreads();

      if (khalf == 0) {
        float v = partial[tid] + partial[tid + 1] + bias[l * HID + row];
        if (l == 0) v += Z0[((size_t)gate * T_SEQ + t) * HID + row];
        pre[rlo][gate] = v;
      }
      __syncthreads();

      if (tid < 32) {
        int r = wg * 32 + tid;
        float f = 1.f / (1.f + __expf(-pre[tid][0]));
        float g = 1.f / (1.f + __expf(-pre[tid][1]));
        float q = 1.f / (1.f + __expf(-pre[tid][2]));
        float c = 1.f / (1.f + __expf(-pre[tid][3]));   // sigmoid candidate, per reference
        float s = f * sstate[l * HID + r] + g * c;
        sstate[l * HID + r] = s;
        hstate[l * HID + r] = tanhf(s) * q;
      }

      // global barrier across all NWG workgroups
      __syncthreads();
      if (tid == 0) {
        __threadfence();
        u32 myg = __hip_atomic_load(gen, __ATOMIC_ACQUIRE, __HIP_MEMORY_SCOPE_AGENT);
        u32 arr = __hip_atomic_fetch_add(cnt, 1u, __ATOMIC_ACQ_REL, __HIP_MEMORY_SCOPE_AGENT);
        if (arr == NWG - 1) {
          __hip_atomic_store(cnt, 0u, __ATOMIC_RELAXED, __HIP_MEMORY_SCOPE_AGENT);
          __hip_atomic_fetch_add(gen, 1u, __ATOMIC_RELEASE, __HIP_MEMORY_SCOPE_AGENT);
        } else {
          while (__hip_atomic_load(gen, __ATOMIC_ACQUIRE, __HIP_MEMORY_SCOPE_AGENT) == myg) {
            __builtin_amdgcn_s_sleep(1);
          }
        }
      }
      __syncthreads();
    }
  }
}

__global__ void k_final(const float* __restrict__ hstate, float* __restrict__ out) {
  int i = blockIdx.x * blockDim.x + threadIdx.x;
  if (i < NLAY * HID) out[i] = hstate[i];
}

extern "C" void kernel_launch(void* const* d_in, const int* in_sizes, int n_in,
                              void* d_out, int out_size, void* d_ws, size_t ws_size,
                              hipStream_t stream) {
  (void)in_sizes; (void)n_in; (void)out_size; (void)ws_size;
  const float* x   = (const float*)d_in[0];
  const float* Uf  = (const float*)d_in[1];
  const float* UHf = (const float*)d_in[2];
  const float* Wf  = (const float*)d_in[3];
  const float* Bf  = (const float*)d_in[4];
  const float* Ug  = (const float*)d_in[5];
  const float* UHg = (const float*)d_in[6];
  const float* Wg  = (const float*)d_in[7];
  const float* Bg  = (const float*)d_in[8];
  const float* Uq  = (const float*)d_in[9];
  const float* UHq = (const float*)d_in[10];
  const float* Wq  = (const float*)d_in[11];
  const float* Bq  = (const float*)d_in[12];
  const float* Uc  = (const float*)d_in[13];
  const float* UHc = (const float*)d_in[14];
  const float* Wc  = (const float*)d_in[15];
  const float* Bc  = (const float*)d_in[16];

  char* base = (char*)d_ws;
  size_t off = 0;
  auto alignup = [](size_t v) { return (v + 255) & ~(size_t)255; };
  u32* syncp = (u32*)(base + off); off += alignup(64 * sizeof(u32));
  u16* xbf   = (u16*)(base + off); off += alignup((size_t)T_SEQ * HID * 2);  // 4 MB
  u16* U4    = (u16*)(base + off); off += alignup((size_t)4  * HID * HID * 2); // 8 MB
  u16* W4    = (u16*)(base + off); off += alignup((size_t)16 * HID * HID * 2); // 32 MB
  u16* UH4   = (u16*)(base + off); off += alignup((size_t)12 * HID * HID * 2); // 24 MB
  float* Z0  = (float*)(base + off); off += alignup((size_t)4 * T_SEQ * HID * 4); // 32 MB
  float* hst = (float*)(base + off); off += alignup((size_t)NLAY * HID * 4);
  float* sst = (float*)(base + off); off += alignup((size_t)NLAY * HID * 4);

  k_init<<<dim3(32), dim3(TPB), 0, stream>>>(syncp, hst, sst);

  const int HH = HID * HID;
  k_cvt<<<dim3((T_SEQ * HID + TPB - 1) / TPB), dim3(TPB), 0, stream>>>(x, xbf, 0, 1, T_SEQ * HID);
  const float* Us[4]  = {Uf, Ug, Uq, Uc};
  const float* Ws[4]  = {Wf, Wg, Wq, Wc};
  const float* UHs[4] = {UHf, UHg, UHq, UHc};
  for (int g = 0; g < 4; ++g) {
    k_cvt<<<dim3((HH + TPB - 1) / TPB), dim3(TPB), 0, stream>>>(Us[g], U4, g, 1, HH);
    k_cvt<<<dim3((4 * HH + TPB - 1) / TPB), dim3(TPB), 0, stream>>>(Ws[g], W4, g, 4, HH);
    k_cvt<<<dim3((3 * HH + TPB - 1) / TPB), dim3(TPB), 0, stream>>>(UHs[g], UH4, g, 3, HH);
  }

  k_z0gemm<<<dim3((256 * 128) / 8), dim3(TPB), 0, stream>>>(U4, xbf, Z0);

  k_scan<<<dim3(NWG), dim3(TPB), 0, stream>>>(W4, UH4, Z0, Bf, Bg, Bq, Bc, hst, sst, syncp);

  k_final<<<dim3((NLAY * HID + TPB - 1) / TPB), dim3(TPB), 0, stream>>>(hst, (float*)d_out);
}
